// conditionalNF_75711683494093
// MI455X (gfx1250) — compile-verified, matrix operand order checked by reading, not run on hardware
//
#include <hip/hip_runtime.h>
#include <hip/hip_bf16.h>
#include <math.h>

// ---------------------------------------------------------------------------
// Problem constants: B=512, F=16, Z=P=256, NF=512, H=512, IN=512, 9 layers
// ---------------------------------------------------------------------------
#define ZD     256
#define FD     16
#define BD     512
#define NROWS  8192        // B*F
#define NFD    512
#define HD     512
#define IND    512
#define LOG2PI 1.8378770664093453f

typedef __attribute__((ext_vector_type(16))) _Float16 v16h;
typedef __attribute__((ext_vector_type(8)))  _Float16 v8h;
typedef __attribute__((ext_vector_type(8)))  float    v8f;

__device__ __forceinline__ float sigmoidf_(float x) { return 1.0f / (1.0f + expf(-x)); }

// MASKS[i]: i==0 -> zeros ('c'); i odd -> odd channels=1 ('a'); i even -> even=1 ('b')
__device__ __forceinline__ float maskval(int layer, int c) {
  if (layer == 0) return 0.0f;
  int bit = (layer & 1) ? (c & 1) : ((c & 1) ^ 1);
  return bit ? 1.0f : 0.0f;
}

// 4 consecutive 16B async global->LDS copies (64B per lane).
// IOFFSET is added to BOTH the LDS and global addresses (ISA 15.18 async),
// which matches contiguous chunks on both sides.
__device__ __forceinline__ void async_cp64(unsigned lds, unsigned gofs, const _Float16* base)
{
  asm volatile(
      "global_load_async_to_lds_b128 %0, %1, %2 offset:0\n\t"
      "global_load_async_to_lds_b128 %0, %1, %2 offset:16\n\t"
      "global_load_async_to_lds_b128 %0, %1, %2 offset:32\n\t"
      "global_load_async_to_lds_b128 %0, %1, %2 offset:48"
      :: "v"(lds), "v"(gofs), "s"(base) : "memory");
}

__device__ __forceinline__ void wait_async0()
{
  asm volatile("s_wait_asynccnt 0x0" ::: "memory");
}

// ---------------------------------------------------------------------------
// Weight preparation: f32 -> f16, all weights end up (N,K) row-major
// ---------------------------------------------------------------------------
// (L,K,N) f32 -> (L,N,K) f16, tiled 32x32 transpose through LDS
__global__ __launch_bounds__(256)
void k_wT(const float* __restrict__ W, _Float16* __restrict__ O, int K, int N)
{
  __shared__ float t[32][33];
  const int l = blockIdx.z;
  const float* Wl = W + (size_t)l * K * N;
  _Float16*    Ol = O + (size_t)l * K * N;
  const int n0 = blockIdx.x * 32, k0 = blockIdx.y * 32;
  const int x = threadIdx.x & 31, y = threadIdx.x >> 5;   // y in 0..7
#pragma unroll
  for (int j = 0; j < 4; ++j)
    t[y + 8 * j][x] = Wl[(size_t)(k0 + y + 8 * j) * N + (n0 + x)];
  __syncthreads();
#pragma unroll
  for (int j = 0; j < 4; ++j)
    Ol[(size_t)(n0 + y + 8 * j) * K + (k0 + x)] = (_Float16)t[x][y + 8 * j];
}

// straight f32 -> f16 convert (for weights already stored (N,K))
__global__ __launch_bounds__(256)
void k_cvt(const float* __restrict__ in, _Float16* __restrict__ out, int n)
{
  int i = blockIdx.x * 256 + threadIdx.x;
  if (i < n) out[i] = (_Float16)in[i];
}

// ---------------------------------------------------------------------------
// WMMA GEMM:  C[M,N] = act( A[M,K] @ W^T + bias )  (optionally += into f32 C)
//   A: f16 (M,K) row-major with row stride lda (halves)
//   W: f16 (N,K) row-major  (pre-transposed/converted)
// Block: 128 threads (4 waves, 2x2). Block tile 128(M)x64(N); wave tile
// 64x32 = 4x2 accumulators. K-step 64, double-buffered LDS.
// Staging uses CDNA5 GLOBAL_LOAD_ASYNC_TO_LDS_B128 (ASYNCcnt): zero staging
// VGPRs, so no scratch spills; copies for tile s+1 overlap the 16 WMMAs on
// tile s, then s_wait_asynccnt 0 + barrier.
// M multiple of 128, N of 64, K of 64.
// ---------------------------------------------------------------------------
#define TK   64
#define LDST 72   // padded LDS row stride (halves), rows stay 16B aligned (144B)

template<int ACT, bool ACC, bool OF16>
__global__ __launch_bounds__(128) __attribute__((amdgpu_waves_per_eu(1)))
void k_gemm(const _Float16* __restrict__ A, const _Float16* __restrict__ W,
            const float* __restrict__ bias, void* __restrict__ Cv,
            int M, int N, int K, int lda)
{
  __shared__ _Float16 As[2][128 * LDST];
  __shared__ _Float16 Bs[2][64 * LDST];

  const int tid   = threadIdx.x;       // 0..127
  const int lane  = tid & 31;
  const int wave  = tid >> 5;          // 0..3
  const int waveM = wave >> 1;         // 0..1
  const int waveN = wave & 1;          // 0..1
  const int m0 = blockIdx.y * 128;
  const int n0 = blockIdx.x * 64;

  // staging: each thread owns 32 consecutive halves (64B) of a row.
  // A tile 128x64: rows srow and srow+64.  B tile 64x64: row srow.
  const int srow = tid >> 1;           // 0..63
  const int skh  = (tid & 1) * 32;

  // LDS byte addresses (workgroup-relative; low 32 bits of generic pointer)
  unsigned lA0[2], lA1[2], lB[2];
#pragma unroll
  for (int b = 0; b < 2; ++b) {
    lA0[b] = (unsigned)(size_t)&As[b][srow * LDST + skh];
    lA1[b] = (unsigned)(size_t)&As[b][(srow + 64) * LDST + skh];
    lB[b]  = (unsigned)(size_t)&Bs[b][srow * LDST + skh];
  }
  // global byte offsets relative to A / W base pointers
  const unsigned oA0 = (unsigned)(((size_t)(m0 + srow)      * lda + skh) * 2);
  const unsigned oA1 = (unsigned)(((size_t)(m0 + srow + 64) * lda + skh) * 2);
  const unsigned oB  = (unsigned)(((size_t)(n0 + srow)      * K   + skh) * 2);

  v8f acc[4][2];
#pragma unroll
  for (int mi = 0; mi < 4; ++mi)
#pragma unroll
    for (int ni = 0; ni < 2; ++ni)
      acc[mi][ni] = (v8f){};

  // prologue: async-stage tile 0 into buffer 0
  async_cp64(lA0[0], oA0, A);
  async_cp64(lA1[0], oA1, A);
  async_cp64(lB[0],  oB,  W);
  wait_async0();
  __syncthreads();

  const int nsteps = K / TK;
  for (int s = 0; s < nsteps; ++s) {
    const int cur = s & 1;
    // issue async copies for the next tile; they overlap the WMMAs below
    if (s + 1 < nsteps) {
      const unsigned kofs = (unsigned)(s + 1) * (TK * 2);
      async_cp64(lA0[cur ^ 1], oA0 + kofs, A);
      async_cp64(lA1[cur ^ 1], oA1 + kofs, A);
      async_cp64(lB[cur ^ 1],  oB  + kofs, W);
    }
    // compute on buffer `cur`
    const int koff = (lane < 16) ? 0 : 8;   // ISA 16-bit A-matrix K layout
#pragma unroll
    for (int kk = 0; kk < TK; kk += 32) {
      v16h af[4], bf[2];
#pragma unroll
      for (int mi = 0; mi < 4; ++mi) {
        int r = waveM * 64 + mi * 16 + (lane & 15);
        v8h lo = *(const v8h*)&As[cur][r * LDST + kk + koff];
        v8h hi = *(const v8h*)&As[cur][r * LDST + kk + koff + 16];
#pragma unroll
        for (int j = 0; j < 8; ++j) { af[mi][j] = lo[j]; af[mi][j + 8] = hi[j]; }
      }
#pragma unroll
      for (int ni = 0; ni < 2; ++ni) {
        int n = waveN * 32 + ni * 16 + (lane & 15);
        v8h lo = *(const v8h*)&Bs[cur][n * LDST + kk + koff];
        v8h hi = *(const v8h*)&Bs[cur][n * LDST + kk + koff + 16];
#pragma unroll
        for (int j = 0; j < 8; ++j) { bf[ni][j] = lo[j]; bf[ni][j + 8] = hi[j]; }
      }
#pragma unroll
      for (int mi = 0; mi < 4; ++mi)
#pragma unroll
        for (int ni = 0; ni < 2; ++ni)
          acc[mi][ni] = __builtin_amdgcn_wmma_f32_16x16x32_f16(
              false, af[mi], false, bf[ni], (short)0, acc[mi][ni], false, false);
    }
    // next tile's async copies must be LDS-visible before the next iteration
    wait_async0();
    __syncthreads();
  }

  // epilogue: D layout = lanes 0-15: N=lane, M=v ; lanes 16-31: N=lane-16, M=8+v
#pragma unroll
  for (int mi = 0; mi < 4; ++mi) {
#pragma unroll
    for (int ni = 0; ni < 2; ++ni) {
      int n  = n0 + waveN * 32 + ni * 16 + (lane & 15);
      int mb = m0 + waveM * 64 + mi * 16 + ((lane >> 4) << 3);
      float bv = bias[n];
#pragma unroll
      for (int v = 0; v < 8; ++v) {
        int m = mb + v;
        float val = acc[mi][ni][v] + bv;
        if (ACC) val += ((const float*)Cv)[(size_t)m * N + n];
        if (ACT == 1) val = (val > 0.0f) ? val : 0.01f * val;   // leaky_relu(0.01)
        if (ACT == 2) val = tanhf(val);
        if (OF16) ((_Float16*)Cv)[(size_t)m * N + n] = (_Float16)val;
        else      ((float*)Cv)[(size_t)m * N + n] = val;
      }
    }
  }
}

// ---------------------------------------------------------------------------
// Flow elementwise kernels
// ---------------------------------------------------------------------------
__global__ __launch_bounds__(256)
void k_init_z(const float* __restrict__ nf, float* __restrict__ z, _Float16* __restrict__ zh)
{
  int idx = blockIdx.x * 256 + threadIdx.x;     // NROWS*ZD
  int row = idx >> 8, c = idx & 255;
  float v = nf[(size_t)row * (2 * ZD) + ZD + c];
  z[idx] = v;
  zh[idx] = (_Float16)v;
}

// inp[row, 0:Z] = (layer==0) ? z_shift : mask*z ;  inp[row, Z:2Z] = p   (f16)
__global__ __launch_bounds__(256)
void k_build_inp(const float* __restrict__ nf, const float* __restrict__ z,
                 _Float16* __restrict__ inp, int layer)
{
  int idx = blockIdx.x * 256 + threadIdx.x;     // NROWS*2Z
  int row = idx >> 9, col = idx & 511;
  float v;
  if (col < ZD) {
    if (layer == 0) {
      int f = row & (FD - 1);
      v = (f == 0) ? 0.0f : z[(size_t)(row - 1) * ZD + col];
    } else {
      v = maskval(layer, col) * z[(size_t)row * ZD + col];
    }
  } else {
    v = nf[(size_t)row * (2 * ZD) + (col - ZD)];   // p = nf_input[:,:,:Z]
  }
  inp[idx] = (_Float16)v;
}

// z = (1-m)*(z - t)*exp(-s) + m*z ;  log_det[row] -= sum_c s ; zh = (f16)z
__global__ __launch_bounds__(256)
void k_update(const float* __restrict__ s_raw, const float* __restrict__ t_raw,
              float* __restrict__ z, _Float16* __restrict__ zh,
              float* __restrict__ log_det, int layer)
{
  int row = blockIdx.x, c = threadIdx.x;
  size_t idx = (size_t)row * ZD + c;
  float m  = maskval(layer, c);
  float im = 1.0f - m;
  float s  = im * s_raw[idx];          // tanh applied in GEMM epilogue
  float t  = im * t_raw[idx];
  float zc = z[idx];
  float zn = im * (zc - t) * expf(-s) + m * zc;
  z[idx]  = zn;
  zh[idx] = (_Float16)zn;

  __shared__ float red[256];
  red[c] = s; __syncthreads();
  for (int o = 128; o > 0; o >>= 1) { if (c < o) red[c] += red[c + o]; __syncthreads(); }
  if (c == 0) log_det[row] -= red[0];
}

// ---------------------------------------------------------------------------
// LSTM prior kernels (h stored f16 since it is only ever a GEMM input)
// ---------------------------------------------------------------------------
__global__ __launch_bounds__(256)
void k_lstm_cell(const float* __restrict__ g, _Float16* __restrict__ h, float* __restrict__ c)
{
  int idx = blockIdx.x * 256 + threadIdx.x;     // B*H
  int b = idx >> 9, j = idx & 511;
  const float* gr = g + (size_t)b * 4 * HD;
  float ig = sigmoidf_(gr[j]);
  float fg = sigmoidf_(gr[HD + j]);
  float gc = tanhf(gr[2 * HD + j]);
  float og = sigmoidf_(gr[3 * HD + j]);
  float cn = fg * c[idx] + ig * gc;
  c[idx] = cn;
  h[idx] = (_Float16)(og * tanhf(cn));
}

// partial[f*B + b] = sum_c [ -0.5*samp^2 - 0.5*log(2pi) - logvar ]
__global__ __launch_bounds__(256)
void k_prior_step(const float* __restrict__ z, const float* __restrict__ mean,
                  const float* __restrict__ logvar, float* __restrict__ partial, int f)
{
  int b = blockIdx.x, c = threadIdx.x;
  size_t mi = (size_t)b * ZD + c;
  float lv = logvar[mi];
  float zp = z[((size_t)b * FD + f) * ZD + c];
  float samp = (zp - mean[mi]) * expf(-lv);
  float val  = -0.5f * samp * samp - 0.5f * LOG2PI - lv;

  __shared__ float red[256];
  red[c] = val; __syncthreads();
  for (int o = 128; o > 0; o >>= 1) { if (c < o) red[c] += red[c + o]; __syncthreads(); }
  if (c == 0) partial[f * BD + b] = red[0];
}

// ---------------------------------------------------------------------------
// Output: z -> d_out[1..], per-row std-normal logprob partials, final scalar
// ---------------------------------------------------------------------------
__global__ __launch_bounds__(256)
void k_zfinal(const float* __restrict__ z, float* __restrict__ out, float* __restrict__ zpart)
{
  int row = blockIdx.x, c = threadIdx.x;
  size_t idx = (size_t)row * ZD + c;
  float zi = z[idx];
  out[1 + idx] = zi;
  float lp = -0.5f * zi * zi - 0.5f * LOG2PI;
  __shared__ float red[256];
  red[c] = lp; __syncthreads();
  for (int o = 128; o > 0; o >>= 1) { if (c < o) red[c] += red[c + o]; __syncthreads(); }
  if (c == 0) zpart[row] = red[0];
}

__global__ __launch_bounds__(256)
void k_loss(const float* __restrict__ log_det, const float* __restrict__ zpart,
            const float* __restrict__ partial, float* __restrict__ out)
{
  int t = threadIdx.x;
  double a = 0.0;
  for (int i = t; i < NROWS; i += 256)
    a += (double)log_det[i] + (double)zpart[i] + 0.0025 * (double)partial[i];
  __shared__ double red[256];
  red[t] = a; __syncthreads();
  for (int o = 128; o > 0; o >>= 1) { if (t < o) red[t] += red[t + o]; __syncthreads(); }
  if (t == 0) out[0] = (float)(-red[0] / (double)BD);
}

// ---------------------------------------------------------------------------
// Host side
// ---------------------------------------------------------------------------
static inline void gemm(int act, bool accum, bool of16,
                        const _Float16* A, const _Float16* W, const float* bias, void* C,
                        int M, int N, int K, int lda, hipStream_t s)
{
  dim3 g(N / 64, M / 128), b(128);
  if (of16) {
    k_gemm<1, false, true><<<g, b, 0, s>>>(A, W, bias, C, M, N, K, lda);      // leaky, f16 out
  } else if (accum) {
    k_gemm<0, true, false><<<g, b, 0, s>>>(A, W, bias, C, M, N, K, lda);      // += , f32 out
  } else if (act == 2) {
    k_gemm<2, false, false><<<g, b, 0, s>>>(A, W, bias, C, M, N, K, lda);     // tanh, f32 out
  } else {
    k_gemm<0, false, false><<<g, b, 0, s>>>(A, W, bias, C, M, N, K, lda);     // none, f32 out
  }
}

extern "C" void kernel_launch(void* const* d_in, const int* in_sizes, int n_in,
                              void* d_out, int out_size, void* d_ws, size_t ws_size,
                              hipStream_t stream)
{
  (void)in_sizes; (void)n_in; (void)out_size; (void)ws_size;

  const float* nf   = (const float*)d_in[0];
  const float* sW1  = (const float*)d_in[1];
  const float* sb1  = (const float*)d_in[2];
  const float* sW2  = (const float*)d_in[3];
  const float* sb2  = (const float*)d_in[4];
  const float* sW3  = (const float*)d_in[5];
  const float* sb3  = (const float*)d_in[6];
  const float* tW1  = (const float*)d_in[7];
  const float* tb1  = (const float*)d_in[8];
  const float* tW2  = (const float*)d_in[9];
  const float* tb2  = (const float*)d_in[10];
  const float* tW3  = (const float*)d_in[11];
  const float* tb3  = (const float*)d_in[12];
  const float* Wih1 = (const float*)d_in[13];
  const float* Whh1 = (const float*)d_in[14];
  const float* bih1 = (const float*)d_in[15];
  const float* bhh1 = (const float*)d_in[16];
  const float* Wih2 = (const float*)d_in[17];
  const float* Whh2 = (const float*)d_in[18];
  const float* bih2 = (const float*)d_in[19];
  const float* bhh2 = (const float*)d_in[20];
  const float* Wm   = (const float*)d_in[21];
  const float* bm   = (const float*)d_in[22];
  const float* Wv   = (const float*)d_in[23];
  const float* bv   = (const float*)d_in[24];

  float* out = (float*)d_out;

  // ---- workspace bump allocation (bytes) ----
  char* p = (char*)d_ws;
  auto alloc = [&](size_t bytes) { char* r = p; p += (bytes + 255) & ~(size_t)255; return (void*)r; };

  // f16 weights, all (N,K) row-major
  _Float16* sW1T  = (_Float16*)alloc((size_t)9 * IND * NFD * 2);
  _Float16* sW2T  = (_Float16*)alloc((size_t)9 * NFD * NFD * 2);
  _Float16* sW3T  = (_Float16*)alloc((size_t)9 * NFD * ZD  * 2);
  _Float16* tW1T  = (_Float16*)alloc((size_t)9 * IND * NFD * 2);
  _Float16* tW2T  = (_Float16*)alloc((size_t)9 * NFD * NFD * 2);
  _Float16* tW3T  = (_Float16*)alloc((size_t)9 * NFD * ZD  * 2);
  _Float16* Wih1h = (_Float16*)alloc((size_t)4 * HD * ZD * 2);
  _Float16* Whh1h = (_Float16*)alloc((size_t)4 * HD * HD * 2);
  _Float16* Wih2h = (_Float16*)alloc((size_t)4 * HD * HD * 2);
  _Float16* Whh2h = (_Float16*)alloc((size_t)4 * HD * HD * 2);
  _Float16* Wmh   = (_Float16*)alloc((size_t)ZD * HD * 2);
  _Float16* Wvh   = (_Float16*)alloc((size_t)ZD * HD * 2);
  // activations
  _Float16* inp   = (_Float16*)alloc((size_t)NROWS * IND * 2);
  _Float16* ping  = (_Float16*)alloc((size_t)NROWS * NFD * 2);
  _Float16* pong  = (_Float16*)alloc((size_t)NROWS * NFD * 2);
  float*    sbuf  = (float*)alloc((size_t)NROWS * ZD * 4);
  float*    tbuf  = (float*)alloc((size_t)NROWS * ZD * 4);
  float*    zbuf  = (float*)alloc((size_t)NROWS * ZD * 4);
  _Float16* zh    = (_Float16*)alloc((size_t)NROWS * ZD * 2);
  float*    log_det = (float*)alloc((size_t)NROWS * 4);
  float*    gates1  = (float*)alloc((size_t)BD * 4 * HD * 4);
  float*    gates2  = (float*)alloc((size_t)BD * 4 * HD * 4);
  _Float16* h1    = (_Float16*)alloc((size_t)BD * HD * 2);
  _Float16* h2    = (_Float16*)alloc((size_t)BD * HD * 2);
  float*    c1    = (float*)alloc((size_t)BD * HD * 4);
  float*    c2    = (float*)alloc((size_t)BD * HD * 4);
  float*    meanb = (float*)alloc((size_t)BD * ZD * 4);
  float*    logvb = (float*)alloc((size_t)BD * ZD * 4);
  float*    partial = (float*)alloc((size_t)FD * BD * 4);
  float*    zpart   = (float*)alloc((size_t)NROWS * 4);

  // ---- weight prep: f32 -> f16, (N,K)-major ----
  k_wT<<<dim3(NFD / 32, IND / 32, 9), 256, 0, stream>>>(sW1, sW1T, IND, NFD);
  k_wT<<<dim3(NFD / 32, NFD / 32, 9), 256, 0, stream>>>(sW2, sW2T, NFD, NFD);
  k_wT<<<dim3(ZD  / 32, NFD / 32, 9), 256, 0, stream>>>(sW3, sW3T, NFD, ZD);
  k_wT<<<dim3(NFD / 32, IND / 32, 9), 256, 0, stream>>>(tW1, tW1T, IND, NFD);
  k_wT<<<dim3(NFD / 32, NFD / 32, 9), 256, 0, stream>>>(tW2, tW2T, NFD, NFD);
  k_wT<<<dim3(ZD  / 32, NFD / 32, 9), 256, 0, stream>>>(tW3, tW3T, NFD, ZD);
  k_cvt<<<(4 * HD * ZD) / 256, 256, 0, stream>>>(Wih1, Wih1h, 4 * HD * ZD);
  k_cvt<<<(4 * HD * HD) / 256, 256, 0, stream>>>(Whh1, Whh1h, 4 * HD * HD);
  k_cvt<<<(4 * HD * HD) / 256, 256, 0, stream>>>(Wih2, Wih2h, 4 * HD * HD);
  k_cvt<<<(4 * HD * HD) / 256, 256, 0, stream>>>(Whh2, Whh2h, 4 * HD * HD);
  k_cvt<<<(ZD * HD) / 256, 256, 0, stream>>>(Wm, Wmh, ZD * HD);
  k_cvt<<<(ZD * HD) / 256, 256, 0, stream>>>(Wv, Wvh, ZD * HD);

  hipMemsetAsync(log_det, 0, NROWS * sizeof(float), stream);
  hipMemsetAsync(h1, 0, BD * HD * sizeof(_Float16), stream);
  hipMemsetAsync(h2, 0, BD * HD * sizeof(_Float16), stream);
  hipMemsetAsync(c1, 0, BD * HD * sizeof(float), stream);
  hipMemsetAsync(c2, 0, BD * HD * sizeof(float), stream);

  // z0 = nf_input[:,:,Z:]
  k_init_z<<<NROWS * ZD / 256, 256, 0, stream>>>(nf, zbuf, zh);

  // ---------------- flow inverse: layers 8 .. 0 ----------------
  for (int i = 8; i >= 0; --i) {
    k_build_inp<<<NROWS * IND / 256, 256, 0, stream>>>(nf, zbuf, inp, i);

    // s-MLP: 512 ->(leaky) 512 ->(leaky) 512 ->(tanh) 256
    gemm(1, false, true,  inp,  sW1T + (size_t)i * IND * NFD, sb1 + i * NFD, ping, NROWS, NFD, IND, IND, stream);
    gemm(1, false, true,  ping, sW2T + (size_t)i * NFD * NFD, sb2 + i * NFD, pong, NROWS, NFD, NFD, NFD, stream);
    gemm(2, false, false, pong, sW3T + (size_t)i * NFD * ZD,  sb3 + i * ZD,  sbuf, NROWS, ZD,  NFD, NFD, stream);
    // t-MLP: same, no final tanh
    gemm(1, false, true,  inp,  tW1T + (size_t)i * IND * NFD, tb1 + i * NFD, ping, NROWS, NFD, IND, IND, stream);
    gemm(1, false, true,  ping, tW2T + (size_t)i * NFD * NFD, tb2 + i * NFD, pong, NROWS, NFD, NFD, NFD, stream);
    gemm(0, false, false, pong, tW3T + (size_t)i * NFD * ZD,  tb3 + i * ZD,  tbuf, NROWS, ZD,  NFD, NFD, stream);

    k_update<<<NROWS, 256, 0, stream>>>(sbuf, tbuf, zbuf, zh, log_det, i);
  }

  // ---------------- LSTM prior over F=16 steps ----------------
  for (int f = 0; f < FD; ++f) {
    const _Float16* zt = zh + (size_t)f * ZD;          // row b at stride F*Z
    gemm(0, false, false, zt, Wih1h, bih1, gates1, BD, 4 * HD, ZD, FD * ZD, stream);
    gemm(0, true,  false, h1, Whh1h, bhh1, gates1, BD, 4 * HD, HD, HD,      stream);
    k_lstm_cell<<<BD * HD / 256, 256, 0, stream>>>(gates1, h1, c1);
    gemm(0, false, false, h1, Wih2h, bih2, gates2, BD, 4 * HD, HD, HD, stream);
    gemm(0, true,  false, h2, Whh2h, bhh2, gates2, BD, 4 * HD, HD, HD, stream);
    k_lstm_cell<<<BD * HD / 256, 256, 0, stream>>>(gates2, h2, c2);
    gemm(0, false, false, h2, Wmh, bm, meanb, BD, ZD, HD, HD, stream);
    gemm(0, false, false, h2, Wvh, bv, logvb, BD, ZD, HD, HD, stream);
    k_prior_step<<<BD, 256, 0, stream>>>(zbuf, meanb, logvb, partial, f);
  }

  // ---------------- outputs ----------------
  k_zfinal<<<NROWS, 256, 0, stream>>>(zbuf, out, zpart);
  k_loss<<<1, 256, 0, stream>>>(log_det, zpart, partial, out);
}
